// Model_34875134444197
// MI455X (gfx1250) — compile-verified, matrix operand order checked by reading, not run on hardware
//
#include <hip/hip_runtime.h>

// ---------------------------------------------------------------------------
// CDNA5 / gfx1250 implementation. wave32.
//  - All GEMMs via v_wmma_f32_16x16x32_bf16 (bf16 in, fp32 accumulate).
//  - Hot GEMMs (k_gemm_fast): guard-free, double-buffered software pipeline;
//    tiles staged global->LDS with global_load_async_to_lds_b64 (ASYNCcnt),
//    slab k+1 issued before computing slab k, drained with s_wait_asynccnt.
//    K padded to a multiple of 32 in the staging buffers (zeros).
//  - Small/odd GEMMs (k_gemm_edge): branch-free clamped loads.
// ---------------------------------------------------------------------------

typedef __attribute__((ext_vector_type(16))) __bf16 v16bf;
typedef __attribute__((ext_vector_type(8)))  float  v8f;
typedef __attribute__((ext_vector_type(4)))  unsigned uint4v;
typedef unsigned short u16;

union FragBF { v16bf v; unsigned int u[8]; };
union U4x16  { uint4v q; u16 s[8]; };

__device__ __forceinline__ u16 f2bf(float f) {
  union { float f; unsigned int u; } x; x.f = f;
  return (u16)((x.u + 0x7FFFu + ((x.u >> 16) & 1u)) >> 16);
}

template<int ACT> __device__ __forceinline__ float act_apply(float v) {
  if (ACT == 1) return fmaxf(v, 0.0f);
  if (ACT == 2) return 1.0f / (1.0f + __expf(-v));
  if (ACT == 3) return tanhf(v);
  return v;
}

#define TBM 64
#define TBN 128
#define TBK 32
#define LDSK 36   // padded row stride (u16); 72B rows keep 8B alignment

// ---------------------------------------------------------------------------
// Shared fragment gather + WMMA step (operates on staged LDS tiles).
// Fragment layout (16-bit 16x32): lanes 0-15 rows 0..15 K{0..7,16..23},
// lanes 16-31 same rows K{8..15,24..31}; pairs contiguous -> dword loads.
// ---------------------------------------------------------------------------
__device__ __forceinline__ void wmma_step(
    const u16 (*As)[LDSK], const u16 (*Bs)[LDSK],
    int wm, int wn, int r, int hl, v8f acc[2][2])
{
  FragBF fa[2], fb[2];
  #pragma unroll
  for (int i = 0; i < 2; ++i) {
    const int mr = wm * 32 + i * 16 + r;
    #pragma unroll
    for (int v = 0; v < 8; ++v) {
      const int k0 = ((v >> 2) << 4) + (hl << 3) + ((v & 3) << 1);
      fa[i].u[v] = *(const unsigned int*)&As[mr][k0];
    }
  }
  #pragma unroll
  for (int j = 0; j < 2; ++j) {
    const int nr = wn * 32 + j * 16 + r;
    #pragma unroll
    for (int v = 0; v < 8; ++v) {
      const int k0 = ((v >> 2) << 4) + (hl << 3) + ((v & 3) << 1);
      fb[j].u[v] = *(const unsigned int*)&Bs[nr][k0];
    }
  }
  #pragma unroll
  for (int i = 0; i < 2; ++i)
    #pragma unroll
    for (int j = 0; j < 2; ++j)
      acc[i][j] = __builtin_amdgcn_wmma_f32_16x16x32_bf16(
          false, fa[i].v, false, fb[j].v, (short)0, acc[i][j], false, false);
}

// ---------------------------------------------------------------------------
// FAST GEMM: preconditions M%64==0, N%128==0, K%32==0, lda%4==0,
//            BT: ldb%4==0 ; NN: ldb%8==0 ; all buffers >=16B aligned.
// Double-buffered: async loads for slab k+1 overlap WMMA on slab k.
// ---------------------------------------------------------------------------
template<int ACT, bool BT>
__global__ __launch_bounds__(256) void k_gemm_fast(
    const u16* __restrict__ A, long lda, long sA,
    const u16* __restrict__ Bm, long ldb, long sB,
    float* __restrict__ C, u16* __restrict__ Cb, long ldc, long sC,
    const float* __restrict__ bias, int M, int N, int K)
{
  __shared__ __align__(16) u16 As[2][TBM][LDSK];
  __shared__ __align__(16) u16 Bs[2][TBN][LDSK];

  const int tid  = threadIdx.x;
  const int row0 = blockIdx.y * TBM;
  const int col0 = blockIdx.x * TBN;
  const long ab = (long)blockIdx.z * sA;
  const long bb = (long)blockIdx.z * sB;
  const long cb = (long)blockIdx.z * sC;

  const int wid = tid >> 5, lane = tid & 31;
  const int wm = wid >> 2, wn = wid & 3;
  const int r = lane & 15, hl = lane >> 4;

  v8f acc[2][2] = {};

  // A staging: thread -> (row, 8 consecutive k); 16B per thread, 2x async b64
  const int arow = tid >> 2, ak0 = (tid & 3) << 3;
  const unsigned a_lds0 = (unsigned)(size_t)&As[0][arow][ak0];
  const unsigned a_lds1 = (unsigned)(size_t)&As[1][arow][ak0];
  const u16* a_gp = A + ab + (long)(row0 + arow) * lda + ak0;
  // B NT staging: thread -> (n, 16 consecutive k); 32B per thread, 4x async b64
  const int bn_t = tid >> 1, bk0_t = (tid & 1) << 4;
  const unsigned bt_lds0 = (unsigned)(size_t)&Bs[0][bn_t][bk0_t];
  const unsigned bt_lds1 = (unsigned)(size_t)&Bs[1][bn_t][bk0_t];
  const u16* bt_gp = Bm + bb + (long)(col0 + bn_t) * ldb + bk0_t;
  // B NN staging: thread -> (k pair, 8 consecutive n); vector load + repack
  const int kp = tid >> 4, n0 = (tid & 15) << 3;
  const u16* bn_gp = Bm + bb + (long)(2 * kp) * ldb + col0 + n0;

  auto stage = [&](int bufi, int kk) {
    const unsigned aL = bufi ? a_lds1 : a_lds0;
    const unsigned long long ga = (unsigned long long)(size_t)(a_gp + kk);
    asm volatile("global_load_async_to_lds_b64 %0, %1, off"
                 :: "v"(aL), "v"(ga) : "memory");
    asm volatile("global_load_async_to_lds_b64 %0, %1, off offset:8"
                 :: "v"(aL), "v"(ga) : "memory");
    if (BT) {
      const unsigned bL = bufi ? bt_lds1 : bt_lds0;
      const unsigned long long gb = (unsigned long long)(size_t)(bt_gp + kk);
      asm volatile("global_load_async_to_lds_b64 %0, %1, off"
                   :: "v"(bL), "v"(gb) : "memory");
      asm volatile("global_load_async_to_lds_b64 %0, %1, off offset:8"
                   :: "v"(bL), "v"(gb) : "memory");
      asm volatile("global_load_async_to_lds_b64 %0, %1, off offset:16"
                   :: "v"(bL), "v"(gb) : "memory");
      asm volatile("global_load_async_to_lds_b64 %0, %1, off offset:24"
                   :: "v"(bL), "v"(gb) : "memory");
    } else {
      // two K-rows (2kp, 2kp+1), 8 consecutive n; repack to packed-K dwords
      const u16* g = bn_gp + (long)kk * ldb;
      U4x16 r0, r1;
      r0.q = *(const uint4v*)g;
      r1.q = *(const uint4v*)(g + ldb);
      u16 (*Bsb)[LDSK] = bufi ? Bs[1] : Bs[0];
      #pragma unroll
      for (int j = 0; j < 8; ++j) {
        const unsigned pv = (unsigned)r0.s[j] | ((unsigned)r1.s[j] << 16);
        *(unsigned*)&Bsb[n0 + j][2 * kp] = pv;
      }
    }
  };

  // software pipeline: stage slab 0; per iteration drain, barrier, issue next,
  // compute current. Buffer nxt was last read in the previous iteration, before this barrier, so
  // overwriting it after this barrier is race-free -> one barrier / iteration.
  stage(0, 0);
  int cur = 0;
  for (int kk = 0; kk < K; kk += TBK) {
    asm volatile("s_wait_asynccnt 0x0" ::: "memory");
    __syncthreads();
    if (kk + TBK < K) stage(cur ^ 1, kk + TBK);
    wmma_step(As[cur], Bs[cur], wm, wn, r, hl, acc);
    cur ^= 1;
  }

  // C/D layout: vgpr i -> (M = i + 8*half, N = lane%16); interior: no guards.
  // Bias hoisted: only two distinct columns per lane.
  float bcol[2] = {0.f, 0.f};
  if (bias) {
    bcol[0] = bias[col0 + wn * 32 + r];
    bcol[1] = bias[col0 + wn * 32 + 16 + r];
  }
  #pragma unroll
  for (int i = 0; i < 2; ++i)
    #pragma unroll
    for (int j = 0; j < 2; ++j) {
      const int colj = col0 + wn * 32 + j * 16 + r;
      long o = cb + (long)(row0 + wm * 32 + i * 16 + hl * 8) * ldc + colj;
      #pragma unroll
      for (int vg = 0; vg < 8; ++vg) {
        float val = act_apply<ACT>(acc[i][j][vg] + bcol[j]);
        if (C)  C[o]  = val;
        if (Cb) Cb[o] = f2bf(val);
        o += ldc;
      }
    }
}

// ---------------------------------------------------------------------------
// EDGE GEMM: generic sizes, branch-free clamped loads. Used only for the
// small batched 38x38 / 80x80 attention GEMMs and the N=2 head.
// ---------------------------------------------------------------------------
template<bool BT>
__global__ __launch_bounds__(256) void k_gemm_edge(
    const u16* __restrict__ A, long lda, long sA,
    const u16* __restrict__ Bm, long ldb, long sB,
    float* __restrict__ C, u16* __restrict__ Cb, long ldc, long sC,
    const float* __restrict__ bias, int M, int N, int K)
{
  __shared__ __align__(16) u16 As[TBM][LDSK];
  __shared__ __align__(16) u16 Bs[TBN][LDSK];

  const int tid  = threadIdx.x;
  const int row0 = blockIdx.y * TBM;
  const int col0 = blockIdx.x * TBN;
  const long ab = (long)blockIdx.z * sA;
  const long bb = (long)blockIdx.z * sB;
  const long cb = (long)blockIdx.z * sC;

  const int wid = tid >> 5, lane = tid & 31;
  const int wm = wid >> 2, wn = wid & 3;
  const int r = lane & 15, hl = lane >> 4;

  v8f acc[2][2] = {};

  const int arow = tid >> 2, ak0 = (tid & 3) << 3;

  for (int kk = 0; kk < K; kk += TBK) {
    {
      const int gr = row0 + arow;
      const int grc = gr < M ? gr : M - 1;
      #pragma unroll
      for (int j = 0; j < 8; ++j) {
        const int gk = kk + ak0 + j;
        const int gkc = gk < K ? gk : K - 1;
        u16 vl = A[ab + (long)grc * lda + gkc];
        if (gr >= M || gk >= K) vl = 0;
        As[arow][ak0 + j] = vl;
      }
    }
    if (BT) {
      const int bn = tid >> 1, bk0 = (tid & 1) << 4;
      const int gn = col0 + bn;
      const int gnc = gn < N ? gn : N - 1;
      #pragma unroll
      for (int j = 0; j < 16; ++j) {
        const int gk = kk + bk0 + j;
        const int gkc = gk < K ? gk : K - 1;
        u16 vl = Bm[bb + (long)gnc * ldb + gkc];
        if (gn >= N || gk >= K) vl = 0;
        Bs[bn][bk0 + j] = vl;
      }
    } else {
      const int bk = tid >> 3, bn0 = (tid & 7) << 4;
      const int gk = kk + bk;
      const int gkc = gk < K ? gk : K - 1;
      #pragma unroll
      for (int j = 0; j < 16; ++j) {
        const int gn = col0 + bn0 + j;
        const int gnc = gn < N ? gn : N - 1;
        u16 vl = Bm[bb + (long)gkc * ldb + gnc];
        if (gn >= N || gk >= K) vl = 0;
        Bs[bn0 + j][bk] = vl;
      }
    }
    __syncthreads();
    wmma_step(As, Bs, wm, wn, r, hl, acc);
    __syncthreads();
  }

  float bcol[2] = {0.f, 0.f};
  if (bias) {
    const int c0 = col0 + wn * 32 + r;
    const int c1 = c0 + 16;
    bcol[0] = bias[c0 < N ? c0 : N - 1];
    bcol[1] = bias[c1 < N ? c1 : N - 1];
  }
  #pragma unroll
  for (int i = 0; i < 2; ++i)
    #pragma unroll
    for (int j = 0; j < 2; ++j) {
      const int colj = col0 + wn * 32 + j * 16 + r;
      const int rbase = row0 + wm * 32 + i * 16 + hl * 8;
      long o = cb + (long)rbase * ldc + colj;
      #pragma unroll
      for (int vg = 0; vg < 8; ++vg) {
        if (rbase + vg < M && colj < N) {
          const float val = acc[i][j][vg] + bcol[j];
          if (C)  C[o]  = val;
          if (Cb) Cb[o] = f2bf(val);
        }
        o += ldc;
      }
    }
}

// ---------------------------------------------------------------------------
// Elementwise / reduction kernels
// ---------------------------------------------------------------------------

// L2-normalize rows of length D (optional pre-relu); bf16 out has row stride
// Dout >= D, zero-padded (so padded-K GEMMs read zeros).
__global__ __launch_bounds__(256) void k_rownorm(
    const float* __restrict__ in, float* __restrict__ outf,
    u16* __restrict__ outb, int D, int Dout, int relu_first)
{
  __shared__ float red[256];
  const long basei = (long)blockIdx.x * D;
  const long baseo = (long)blockIdx.x * Dout;
  const int tid = threadIdx.x;
  float ss = 0.f;
  for (int c = tid; c < D; c += 256) {
    float v = in[basei + c];
    if (relu_first) v = fmaxf(v, 0.f);
    ss += v * v;
  }
  red[tid] = ss; __syncthreads();
  for (int s = 128; s > 0; s >>= 1) {
    if (tid < s) red[tid] += red[tid + s];
    __syncthreads();
  }
  const float sc = 1.0f / fmaxf(sqrtf(red[0]), 1e-12f);
  for (int c = tid; c < D; c += 256) {
    float v = in[basei + c];
    if (relu_first) v = fmaxf(v, 0.f);
    v *= sc;
    if (outf) outf[basei + c] = v;
    if (outb) outb[baseo + c] = f2bf(v);
  }
  if (outb)
    for (int c = D + tid; c < Dout; c += 256) outb[baseo + c] = 0;
}

// row-wise: out = exp(S)*W / sum(exp(S)*W)   (rows of length Mc <= 128)
__global__ __launch_bounds__(128) void k_expnorm(
    const float* __restrict__ S, const float* __restrict__ Wm,
    u16* __restrict__ outb, int Mc)
{
  __shared__ float red[128];
  const long base = (long)blockIdx.x * Mc;
  const int tid = threadIdx.x;
  float v = 0.f;
  if (tid < Mc) v = __expf(S[base + tid]) * Wm[base + tid];
  red[tid] = v; __syncthreads();
  for (int s = 64; s > 0; s >>= 1) {
    if (tid < s) red[tid] += red[tid + s];
    __syncthreads();
  }
  const float inv = 1.0f / red[0];
  if (tid < Mc) outb[base + tid] = f2bf(v * inv);
}

__global__ void k_f2bf(const float* __restrict__ in, u16* __restrict__ out, long n) {
  long i = (long)blockIdx.x * 256 + threadIdx.x;
  if (i < n) out[i] = f2bf(in[i]);
}

// padded convert: dst[r][c] = (r<srows && c<scols) ? bf16(src[r][c]) : 0
__global__ void k_f2bf_pad(const float* __restrict__ src, u16* __restrict__ dst,
                           int srows, int scols, int drows, int dcols) {
  long i = (long)blockIdx.x * 256 + threadIdx.x;
  if (i >= (long)drows * dcols) return;
  int rr = (int)(i / dcols), c = (int)(i % dcols);
  float v = (rr < srows && c < scols) ? src[(long)rr * scols + c] : 0.f;
  dst[i] = f2bf(v);
}

__global__ void k_zero_f(float* __restrict__ p, long n) {
  long i = (long)blockIdx.x * 256 + threadIdx.x;
  if (i < n) p[i] = 0.f;
}
__global__ void k_zero_h(u16* __restrict__ p, long n) {
  long i = (long)blockIdx.x * 256 + threadIdx.x;
  if (i < n) p[i] = 0;
}

// tok = bf16(concat(w, tag_tab[p], zero-pad))  rows = B*L, cols = 2176
__global__ void k_tok(const float* __restrict__ w, const int* __restrict__ p,
                      const float* __restrict__ tag, u16* __restrict__ tok, long n) {
  long i = (long)blockIdx.x * 256 + threadIdx.x;
  if (i >= n) return;
  long row = i / 2176; int c = (int)(i % 2176);
  float v = 0.f;
  if (c < 2048)      v = w[row * 2048 + c];
  else if (c < 2148) v = tag[(long)p[row] * 100 + (c - 2048)];
  tok[i] = f2bf(v);
}

__global__ void k_posgather(const float* __restrict__ tab, const int* __restrict__ ix,
                            float* __restrict__ out, long n) {
  long i = (long)blockIdx.x * 256 + threadIdx.x;
  if (i >= n) return;
  long row = i / 300; int c = (int)(i % 300);
  out[i] = tab[(long)ix[row] * 300 + c];
}

// cat[row, 2048 + c] = bf16(pw2[b, idx(row), c]);  idx==0 -> PL-1
__global__ void k_newpw(const float* __restrict__ pw2, const int* __restrict__ ind,
                        u16* __restrict__ cat, int L, int PL, long n) {
  long i = (long)blockIdx.x * 256 + threadIdx.x;
  if (i >= n) return;
  long row = i / 512; int c = (int)(i % 512);
  long b = row / L;
  int idx = ind[row]; if (idx == 0) idx = PL - 1;
  float v = pw2[((b * PL) + idx) * 512L + c];
  cat[row * 2560 + 2048 + c] = f2bf(v);
}

// one GRU timestep (PyTorch gate order r,z,n); writes h (f32+bf16) and y (bf16)
__global__ __launch_bounds__(256) void k_gru_cell(
    const float* __restrict__ Gi, const float* __restrict__ Gh,
    float* __restrict__ hf, u16* __restrict__ hb,
    u16* __restrict__ y, long ldy, int yoff, int t, int Lp, int H, long n)
{
  long i = (long)blockIdx.x * 256 + threadIdx.x;
  if (i >= n) return;
  long b = i / H; int j = (int)(i % H);
  long gib = ((b * Lp) + t) * (long)(3 * H);
  long ghb = b * (long)(3 * H);
  float rr = 1.f / (1.f + __expf(-(Gi[gib + j] + Gh[ghb + j])));
  float zz = 1.f / (1.f + __expf(-(Gi[gib + H + j] + Gh[ghb + H + j])));
  float nn = tanhf(Gi[gib + 2 * H + j] + rr * Gh[ghb + 2 * H + j]);
  float hn = (1.f - zz) * nn + zz * hf[i];
  hf[i] = hn; hb[i] = f2bf(hn);
  y[((b * Lp) + t) * ldy + yoff + j] = f2bf(hn);
}

// word-graph combine: out = act(h + (adj@h) * sigmoid_gate)
__global__ void k_combine(const float* __restrict__ h, const float* __restrict__ ah,
                          const float* __restrict__ g, float* __restrict__ outf,
                          u16* __restrict__ outb, long n, int relu) {
  long i = (long)blockIdx.x * 256 + threadIdx.x;
  if (i >= n) return;
  float v = h[i] + ah[i] * g[i];
  if (relu) v = fmaxf(v, 0.f);
  if (outf) outf[i] = v;
  if (outb) outb[i] = f2bf(v);
}

// ---------------------------------------------------------------------------
// Host side
// ---------------------------------------------------------------------------
static inline long cdiv(long a, long b) { return (a + b - 1) / b; }

static void launch_gemm(hipStream_t st, bool fast, const u16* A, long lda, long sA,
                        const u16* Bm, long ldb, long sB, bool bT,
                        float* C, u16* Cb, long ldc, long sC,
                        const float* bias, int M, int N, int K, int batch, int act)
{
  dim3 g((unsigned)cdiv(N, TBN), (unsigned)cdiv(M, TBM), (unsigned)batch);
  dim3 b(256);
  if (fast) {
    if (bT) {
      k_gemm_fast<0, true><<<g, b, 0, st>>>(A, lda, sA, Bm, ldb, sB, C, Cb, ldc, sC, bias, M, N, K);
    } else {
      switch (act) {
        case 2:  k_gemm_fast<2, false><<<g, b, 0, st>>>(A, lda, sA, Bm, ldb, sB, C, Cb, ldc, sC, bias, M, N, K); break;
        case 3:  k_gemm_fast<3, false><<<g, b, 0, st>>>(A, lda, sA, Bm, ldb, sB, C, Cb, ldc, sC, bias, M, N, K); break;
        default: k_gemm_fast<0, false><<<g, b, 0, st>>>(A, lda, sA, Bm, ldb, sB, C, Cb, ldc, sC, bias, M, N, K); break;
      }
    }
  } else {
    if (bT) k_gemm_edge<true><<<g, b, 0, st>>>(A, lda, sA, Bm, ldb, sB, C, Cb, ldc, sC, bias, M, N, K);
    else    k_gemm_edge<false><<<g, b, 0, st>>>(A, lda, sA, Bm, ldb, sB, C, Cb, ldc, sC, bias, M, N, K);
  }
}

extern "C" void kernel_launch(void* const* d_in, const int* in_sizes, int n_in,
                              void* d_out, int out_size, void* d_ws, size_t ws_size,
                              hipStream_t stream)
{
  const int B = 512, L = 38, E = 2048, DTOK = 2148, DTOKP = 2176;
  const int H = 1024, G3 = 3072;
  const int PL = 80, PD = 300, PDP = 320, GD = 512, WGD = 512, DCAT = 2560, NC = 2;
  const long BL = (long)B * L;     // 19456
  const long BP = (long)B * PL;    // 40960

  const float* w       = (const float*)d_in[0];
  const int*   p       = (const int*)  d_in[1];
  const float* mask    = (const float*)d_in[2];
  const int*   pos_x   = (const int*)  d_in[3];
  const float* adj     = (const float*)d_in[4];
  const int*   indicate= (const int*)  d_in[5];
  const float* tag_tab = (const float*)d_in[6];
  const float* pos_tab = (const float*)d_in[7];
  const float* wg1_W   = (const float*)d_in[8];  const float* wg1_Wb = (const float*)d_in[9];
  const float* wg1_G   = (const float*)d_in[10]; const float* wg1_Gb = (const float*)d_in[11];
  const float* wg2_W   = (const float*)d_in[12]; const float* wg2_Wb = (const float*)d_in[13];
  const float* wg2_G   = (const float*)d_in[14]; const float* wg2_Gb = (const float*)d_in[15];
  const float* gc1_Wk  = (const float*)d_in[16]; const float* gc1_bk = (const float*)d_in[17];
  const float* gc1_Wv  = (const float*)d_in[18]; const float* gc1_bv = (const float*)d_in[19];
  const float* gc1_Wq  = (const float*)d_in[20]; const float* gc1_bq = (const float*)d_in[21];
  const float* gc2_Wk  = (const float*)d_in[22]; const float* gc2_bk = (const float*)d_in[23];
  const float* gc2_Wv  = (const float*)d_in[24]; const float* gc2_bv = (const float*)d_in[25];
  const float* gc2_Wq  = (const float*)d_in[26]; const float* gc2_bq = (const float*)d_in[27];
  const float* g0f_Wih = (const float*)d_in[28]; const float* g0f_Whh = (const float*)d_in[29];
  const float* g0f_bih = (const float*)d_in[30]; const float* g0f_bhh = (const float*)d_in[31];
  const float* g0b_Wih = (const float*)d_in[32]; const float* g0b_Whh = (const float*)d_in[33];
  const float* g0b_bih = (const float*)d_in[34]; const float* g0b_bhh = (const float*)d_in[35];
  const float* g1f_Wih = (const float*)d_in[36]; const float* g1f_Whh = (const float*)d_in[37];
  const float* g1f_bih = (const float*)d_in[38]; const float* g1f_bhh = (const float*)d_in[39];
  const float* g1b_Wih = (const float*)d_in[40]; const float* g1b_Whh = (const float*)d_in[41];
  const float* g1b_bih = (const float*)d_in[42]; const float* g1b_bhh = (const float*)d_in[43];
  const float* W1_w    = (const float*)d_in[44]; const float* W1_b   = (const float*)d_in[45];
  const float* W2_w    = (const float*)d_in[46]; const float* W2_b   = (const float*)d_in[47];

  float* out_prob = (float*)d_out;
  float* out_wg   = (float*)d_out + BL * NC;

  // --- workspace bump allocator -------------------------------------------
  char* ws = (char*)d_ws;
  size_t off = 0;
  auto alloc = [&](size_t bytes) -> void* {
    void* pt = ws + off;
    off += bytes; off = (off + 255) & ~(size_t)255;
    return pt;
  };

  u16*   wemb_bf = (u16*)  alloc(BL * E * 2);
  float* scr     = (float*)alloc((long)B * 6400 * 4);          // scores (shared)
  u16*   wadj_bf = (u16*)  alloc((long)B * L * L * 2);
  float* buf_h   = (float*)alloc(BL * WGD * 4);
  u16*   buf_hbf = (u16*)  alloc(BL * WGD * 2);
  float* buf_g   = (float*)alloc(BL * WGD * 4);
  float* buf_ah  = (float*)alloc(BL * WGD * 4);
  u16*   wg1_bf  = (u16*)  alloc(BL * WGD * 2);
  u16*   tok_bf  = (u16*)  alloc(BL * DTOKP * 2);
  float* Gi      = (float*)alloc(BL * G3 * 4);
  u16*   hseq_bf = (u16*)  alloc(BL * 2 * H * 2);
  float* hprev   = (float*)alloc((long)B * H * 4);
  u16*   hprevb  = (u16*)  alloc((long)B * H * 2);
  float* Gh      = (float*)alloc((long)B * G3 * 4);
  u16*   cat_bf  = (u16*)  alloc(BL * DCAT * 2);
  float* posw    = (float*)alloc(BP * PD * 4);
  u16*   pw0_bf  = (u16*)  alloc(BP * PDP * 2);
  u16*   q_bf    = (u16*)  alloc(BP * GD * 2);
  u16*   k_bf    = (u16*)  alloc(BP * GD * 2);
  u16*   v_bf    = (u16*)  alloc(BP * GD * 2);
  u16*   wgt_bf  = (u16*)  alloc((long)B * PL * PL * 2);
  float* pw1     = (float*)alloc(BP * GD * 4);
  u16*   pw1n_bf = (u16*)  alloc(BP * GD * 2);
  float* pw2     = (float*)alloc(BP * GD * 4);
  u16*   s2_bf   = (u16*)  alloc(BL * H * 2);
  // weight bf16 mirrors (K-padded where needed)
  u16* wg1W_b = (u16*)alloc((long)E * WGD * 2);
  u16* wg1G_b = (u16*)alloc((long)E * WGD * 2);
  u16* wg2W_b = (u16*)alloc((long)WGD * WGD * 2);
  u16* wg2G_b = (u16*)alloc((long)WGD * WGD * 2);
  u16* g1k_b  = (u16*)alloc((long)PDP * GD * 2);
  u16* g1v_b  = (u16*)alloc((long)PDP * GD * 2);
  u16* g1q_b  = (u16*)alloc((long)PDP * GD * 2);
  u16* g2k_b  = (u16*)alloc((long)GD * GD * 2);
  u16* g2v_b  = (u16*)alloc((long)GD * GD * 2);
  u16* g2q_b  = (u16*)alloc((long)GD * GD * 2);
  u16* g0fWi_b = (u16*)alloc((long)G3 * DTOKP * 2);
  u16* g0fWh_b = (u16*)alloc((long)G3 * H * 2);
  u16* g0bWi_b = (u16*)alloc((long)G3 * DTOKP * 2);
  u16* g0bWh_b = (u16*)alloc((long)G3 * H * 2);
  u16* g1fWi_b = (u16*)alloc((long)G3 * 2 * H * 2);
  u16* g1fWh_b = (u16*)alloc((long)G3 * H * 2);
  u16* g1bWi_b = (u16*)alloc((long)G3 * 2 * H * 2);
  u16* g1bWh_b = (u16*)alloc((long)G3 * H * 2);
  u16* W1_bb   = (u16*)alloc((long)DCAT * H * 2);
  u16* W2_bb   = (u16*)alloc((long)H * NC * 2);
  (void)ws_size; (void)in_sizes; (void)n_in; (void)out_size;

  auto conv = [&](const float* src, u16* dst, long n) {
    k_f2bf<<<(unsigned)cdiv(n, 256), 256, 0, stream>>>(src, dst, n);
  };
  auto convp = [&](const float* src, u16* dst, int sr, int sc, int dr, int dc) {
    k_f2bf_pad<<<(unsigned)cdiv((long)dr * dc, 256), 256, 0, stream>>>(src, dst, sr, sc, dr, dc);
  };

  // --- weight conversions --------------------------------------------------
  conv(wg1_W, wg1W_b, (long)E * WGD);   conv(wg1_G, wg1G_b, (long)E * WGD);
  conv(wg2_W, wg2W_b, (long)WGD * WGD); conv(wg2_G, wg2G_b, (long)WGD * WGD);
  convp(gc1_Wk, g1k_b, PD, GD, PDP, GD);  // K rows padded 300->320
  convp(gc1_Wv, g1v_b, PD, GD, PDP, GD);
  convp(gc1_Wq, g1q_b, PD, GD, PDP, GD);
  conv(gc2_Wk, g2k_b, (long)GD * GD);   conv(gc2_Wv, g2v_b, (long)GD * GD);
  conv(gc2_Wq, g2q_b, (long)GD * GD);
  convp(g0f_Wih, g0fWi_b, G3, DTOK, G3, DTOKP);  // K cols padded 2148->2176
  convp(g0b_Wih, g0bWi_b, G3, DTOK, G3, DTOKP);
  conv(g0f_Whh, g0fWh_b, (long)G3 * H); conv(g0b_Whh, g0bWh_b, (long)G3 * H);
  conv(g1f_Wih, g1fWi_b, (long)G3 * 2 * H); conv(g1f_Whh, g1fWh_b, (long)G3 * H);
  conv(g1b_Wih, g1bWi_b, (long)G3 * 2 * H); conv(g1b_Whh, g1bWh_b, (long)G3 * H);
  conv(W1_w, W1_bb, (long)DCAT * H);    conv(W2_w, W2_bb, (long)H * NC);

  // --- word-graph branch ---------------------------------------------------
  k_rownorm<<<(unsigned)BL, 256, 0, stream>>>(w, nullptr, wemb_bf, E, E, 0);
  // scores = w_emb @ w_emb^T   (batched NT, 38x38 -> edge)
  launch_gemm(stream, false, wemb_bf, E, (long)L * E, wemb_bf, E, (long)L * E, true,
              scr, nullptr, L, (long)L * L, nullptr, L, L, E, B, 0);
  k_expnorm<<<(unsigned)BL, 128, 0, stream>>>(scr, mask, wadj_bf, L);
  // layer 1
  launch_gemm(stream, true, wemb_bf, E, 0, wg1W_b, WGD, 0, false,
              buf_h, buf_hbf, WGD, 0, wg1_Wb, (int)BL, WGD, E, 1, 0);
  launch_gemm(stream, true, wemb_bf, E, 0, wg1G_b, WGD, 0, false,
              buf_g, nullptr, WGD, 0, wg1_Gb, (int)BL, WGD, E, 1, 2 /*sigmoid*/);
  launch_gemm(stream, false, wadj_bf, L, (long)L * L, buf_hbf, WGD, (long)L * WGD, false,
              buf_ah, nullptr, WGD, (long)L * WGD, nullptr, L, WGD, L, B, 0);
  k_combine<<<(unsigned)cdiv(BL * WGD, 256), 256, 0, stream>>>(
      buf_h, buf_ah, buf_g, nullptr, wg1_bf, BL * WGD, 1);
  // layer 2 (output is the model's second return value "wg")
  launch_gemm(stream, true, wg1_bf, WGD, 0, wg2W_b, WGD, 0, false,
              buf_h, buf_hbf, WGD, 0, wg2_Wb, (int)BL, WGD, WGD, 1, 0);
  launch_gemm(stream, true, wg1_bf, WGD, 0, wg2G_b, WGD, 0, false,
              buf_g, nullptr, WGD, 0, wg2_Gb, (int)BL, WGD, WGD, 1, 2);
  launch_gemm(stream, false, wadj_bf, L, (long)L * L, buf_hbf, WGD, (long)L * WGD, false,
              buf_ah, nullptr, WGD, (long)L * WGD, nullptr, L, WGD, L, B, 0);
  k_combine<<<(unsigned)cdiv(BL * WGD, 256), 256, 0, stream>>>(
      buf_h, buf_ah, buf_g, out_wg, nullptr, BL * WGD, 0);

  // --- sentence GRU branch -------------------------------------------------
  k_tok<<<(unsigned)cdiv(BL * DTOKP, 256), 256, 0, stream>>>(w, p, tag_tab, tok_bf, BL * DTOKP);

  auto run_gru = [&](const u16* x_bf, int Din, const u16* Wih_b, const u16* Whh_b,
                     const float* bih, const float* bhh, bool reverse,
                     u16* ybuf, long ldy, int yoff) {
    launch_gemm(stream, true, x_bf, Din, 0, Wih_b, Din, 0, true,
                Gi, nullptr, G3, 0, bih, (int)BL, G3, Din, 1, 0);
    k_zero_f<<<(unsigned)cdiv((long)B * H, 256), 256, 0, stream>>>(hprev, (long)B * H);
    k_zero_h<<<(unsigned)cdiv((long)B * H, 256), 256, 0, stream>>>(hprevb, (long)B * H);
    for (int s = 0; s < L; ++s) {
      int t = reverse ? (L - 1 - s) : s;
      launch_gemm(stream, true, hprevb, H, 0, Whh_b, H, 0, true,
                  Gh, nullptr, G3, 0, bhh, B, G3, H, 1, 0);
      k_gru_cell<<<(unsigned)cdiv((long)B * H, 256), 256, 0, stream>>>(
          Gi, Gh, hprev, hprevb, ybuf, ldy, yoff, t, L, H, (long)B * H);
    }
  };

  run_gru(tok_bf, DTOKP, g0fWi_b, g0fWh_b, g0f_bih, g0f_bhh, false, hseq_bf, 2 * H, 0);
  run_gru(tok_bf, DTOKP, g0bWi_b, g0bWh_b, g0b_bih, g0b_bhh, true,  hseq_bf, 2 * H, H);
  run_gru(hseq_bf, 2 * H, g1fWi_b, g1fWh_b, g1f_bih, g1f_bhh, false, cat_bf, DCAT, 0);
  run_gru(hseq_bf, 2 * H, g1bWi_b, g1bWh_b, g1b_bih, g1b_bhh, true,  cat_bf, DCAT, H);

  // --- POS graph branch ----------------------------------------------------
  k_posgather<<<(unsigned)cdiv(BP * PD, 256), 256, 0, stream>>>(pos_tab, pos_x, posw, BP * PD);
  k_rownorm<<<(unsigned)BP, 256, 0, stream>>>(posw, nullptr, pw0_bf, PD, PDP, 0);
  // layer 1: k,v,q  (K padded 300->320; padded rows of W are zero)
  launch_gemm(stream, true, pw0_bf, PDP, 0, g1k_b, GD, 0, false, nullptr, k_bf, GD, 0, gc1_bk, (int)BP, GD, PDP, 1, 0);
  launch_gemm(stream, true, pw0_bf, PDP, 0, g1v_b, GD, 0, false, nullptr, v_bf, GD, 0, gc1_bv, (int)BP, GD, PDP, 1, 0);
  launch_gemm(stream, true, pw0_bf, PDP, 0, g1q_b, GD, 0, false, nullptr, q_bf, GD, 0, gc1_bq, (int)BP, GD, PDP, 1, 0);
  launch_gemm(stream, false, q_bf, GD, (long)PL * GD, k_bf, GD, (long)PL * GD, true,
              scr, nullptr, PL, (long)PL * PL, nullptr, PL, PL, GD, B, 0);
  k_expnorm<<<(unsigned)BP, 128, 0, stream>>>(scr, adj, wgt_bf, PL);
  launch_gemm(stream, false, wgt_bf, PL, (long)PL * PL, v_bf, GD, (long)PL * GD, false,
              pw1, nullptr, GD, (long)PL * GD, nullptr, PL, GD, PL, B, 0);
  k_rownorm<<<(unsigned)BP, 256, 0, stream>>>(pw1, nullptr, pw1n_bf, GD, GD, 1 /*relu first*/);
  // layer 2
  launch_gemm(stream, true, pw1n_bf, GD, 0, g2k_b, GD, 0, false, nullptr, k_bf, GD, 0, gc2_bk, (int)BP, GD, GD, 1, 0);
  launch_gemm(stream, true, pw1n_bf, GD, 0, g2v_b, GD, 0, false, nullptr, v_bf, GD, 0, gc2_bv, (int)BP, GD, GD, 1, 0);
  launch_gemm(stream, true, pw1n_bf, GD, 0, g2q_b, GD, 0, false, nullptr, q_bf, GD, 0, gc2_bq, (int)BP, GD, GD, 1, 0);
  launch_gemm(stream, false, q_bf, GD, (long)PL * GD, k_bf, GD, (long)PL * GD, true,
              scr, nullptr, PL, (long)PL * PL, nullptr, PL, PL, GD, B, 0);
  k_expnorm<<<(unsigned)BP, 128, 0, stream>>>(scr, adj, wgt_bf, PL);
  launch_gemm(stream, false, wgt_bf, PL, (long)PL * PL, v_bf, GD, (long)PL * GD, false,
              pw2, nullptr, GD, (long)PL * GD, nullptr, PL, GD, PL, B, 0);
  // indicate-gather into cat[:, 2048:2560]
  k_newpw<<<(unsigned)cdiv(BL * GD, 256), 256, 0, stream>>>(pw2, indicate, cat_bf, L, PL, BL * GD);

  // --- head ----------------------------------------------------------------
  launch_gemm(stream, true, cat_bf, DCAT, 0, W1_bb, H, 0, false,
              nullptr, s2_bf, H, 0, W1_b, (int)BL, H, DCAT, 1, 3 /*tanh*/);
  launch_gemm(stream, false, s2_bf, H, 0, W2_bb, NC, 0, false,
              out_prob, nullptr, NC, 0, W2_b, (int)BL, NC, H, 1, 0);
}
